// ASGCNN_pretrain_13194139533625
// MI455X (gfx1250) — compile-verified, matrix operand order checked by reading, not run on hardware
//
#include <hip/hip_runtime.h>

// ---------------------------------------------------------------------------
// ASGCNN forward on MI455X (gfx1250, wave32, WMMA).
// All GEMMs go through one bf16-WMMA kernel (v_wmma_f32_16x16x32_bf16, f32
// accumulate).  Weights are tiny (<=270KB) -> L2 resident.  The streamed A
// operand (the [E,3n] concat matrices) is padded to a 16B-aligned leading
// dimension so the steady-state K loop uses global_load_b128.  Each wave
// computes a 32x16 C tile (2 WMMAs sharing one B fragment).
// ---------------------------------------------------------------------------

typedef __attribute__((ext_vector_type(16))) __bf16 vbf16x16;
typedef __attribute__((ext_vector_type(8)))  float  v8f;

#define ACT_NONE     0
#define ACT_SILU     1
#define ACT_SOFTPLUS 2
#define ACT_ELU      3

__device__ __forceinline__ float act_apply(float x, int act) {
  if (act == ACT_SILU)     return x * (1.0f / (1.0f + __expf(-x)));
  if (act == ACT_SOFTPLUS) return (x > 20.0f) ? x : log1pf(__expf(x));
  if (act == ACT_ELU)      return (x > 0.0f) ? x : (__expf(x) - 1.0f);
  return x;
}

__device__ __forceinline__ vbf16x16 pack16(const float* f) {
  vbf16x16 v;
#pragma unroll
  for (int i = 0; i < 16; ++i) v[i] = (__bf16)f[i];
  return v;
}

// C[M,N] = A[M,K] @ W[K,N] + bias[N]   (bf16 WMMA, f32 accumulate)
// A has row stride lda (>= K).  Block = 256 threads = 8 waves arranged
// 4 (M) x 2 (N); each wave owns a 32x16 C tile (two 16x16 WMMA tiles
// sharing one B fragment) => 128x32 C per block.
//
// OOB handling: an OOB A-row only feeds D-rows that are never stored, and an
// OOB B-column only feeds D-columns that are never stored, so row/col bases
// are clamped (addresses stay legal) and values are NOT masked.  Only the
// K-tail masks values, branchlessly (clamped load + select) with *signed*
// relative offsets so clamped reads stay inside a valid row.
__global__ void gemm_bias_wmma(const float* __restrict__ A, int lda,
                               const float* __restrict__ W,
                               const float* __restrict__ bias,
                               float* __restrict__ C,
                               int M, int N, int K)
{
  const int lane  = threadIdx.x & 31;
  const int wave  = threadIdx.x >> 5;
  const int tileM = (blockIdx.x * 4 + (wave >> 1)) * 32;   // wave-uniform
  const int tileN = (blockIdx.y * 2 + (wave & 1)) * 16;    // wave-uniform
  if (tileM >= M || tileN >= N) return;                    // whole-wave exit

  // ISA 7.12.2 fragment layout:
  //  A 16x32 bf16: lane<16 holds row M=lane, K = {0..7, 16..23};
  //                lane>=16 holds same row, K = {8..15, 24..31}.
  //  B 32x16 bf16: mirrored with lane -> column.
  //  C/D 16x16 f32: lane%16 = column; lanes<16 rows 0..7, lanes>=16 rows 8..15.
  const int half  = lane >> 4;
  const int l16   = lane & 15;
  const int rowLo = tileM + l16;
  const int rowHi = tileM + 16 + l16;
  const int colB  = tileN + l16;
  const int koff  = half * 8;

  // Bases are pre-offset by koff; advanced by 32 (A) / 32*N (B) per K step.
  const float* __restrict__ A0 = A + (size_t)(rowLo < M ? rowLo : M - 1) * lda + koff;
  const float* __restrict__ A1 = A + (size_t)(rowHi < M ? rowHi : M - 1) * lda + koff;
  const float* __restrict__ Wp = W + (colB < N ? colB : N - 1) + (size_t)koff * N;
  const size_t wstep = (size_t)32 * N;

  v8f acc0 = {}, acc1 = {};
  int kt = 0;

  if ((lda & 3) == 0) {
    // Fast path: A segments are 16B aligned -> global_load_b128 x8 per iter.
    for (; kt + 32 <= K; kt += 32) {
      float4 a0 = ((const float4*)A0)[0];
      float4 a1 = ((const float4*)A0)[1];
      float4 a2 = ((const float4*)(A0 + 16))[0];
      float4 a3 = ((const float4*)(A0 + 16))[1];
      float4 c0 = ((const float4*)A1)[0];
      float4 c1 = ((const float4*)A1)[1];
      float4 c2 = ((const float4*)(A1 + 16))[0];
      float4 c3 = ((const float4*)(A1 + 16))[1];
      float bb[16];
#pragma unroll
      for (int i = 0; i < 8; ++i) {
        bb[i]     = Wp[(size_t)i * N];          // loop-invariant offsets
        bb[i + 8] = Wp[(size_t)(i + 16) * N];
      }
      __builtin_prefetch(A0 + 32, 0, 3);        // -> global_prefetch_b8
      __builtin_prefetch(A1 + 32, 0, 3);
      vbf16x16 a, c;
      a[0]  = (__bf16)a0.x; a[1]  = (__bf16)a0.y; a[2]  = (__bf16)a0.z; a[3]  = (__bf16)a0.w;
      a[4]  = (__bf16)a1.x; a[5]  = (__bf16)a1.y; a[6]  = (__bf16)a1.z; a[7]  = (__bf16)a1.w;
      a[8]  = (__bf16)a2.x; a[9]  = (__bf16)a2.y; a[10] = (__bf16)a2.z; a[11] = (__bf16)a2.w;
      a[12] = (__bf16)a3.x; a[13] = (__bf16)a3.y; a[14] = (__bf16)a3.z; a[15] = (__bf16)a3.w;
      c[0]  = (__bf16)c0.x; c[1]  = (__bf16)c0.y; c[2]  = (__bf16)c0.z; c[3]  = (__bf16)c0.w;
      c[4]  = (__bf16)c1.x; c[5]  = (__bf16)c1.y; c[6]  = (__bf16)c1.z; c[7]  = (__bf16)c1.w;
      c[8]  = (__bf16)c2.x; c[9]  = (__bf16)c2.y; c[10] = (__bf16)c2.z; c[11] = (__bf16)c2.w;
      c[12] = (__bf16)c3.x; c[13] = (__bf16)c3.y; c[14] = (__bf16)c3.z; c[15] = (__bf16)c3.w;
      vbf16x16 b = pack16(bb);
      acc0 = __builtin_amdgcn_wmma_f32_16x16x32_bf16(false, a, false, b,
                                                     (short)0, acc0, false, false);
      acc1 = __builtin_amdgcn_wmma_f32_16x16x32_bf16(false, c, false, b,
                                                     (short)0, acc1, false, false);
      A0 += 32; A1 += 32; Wp += wstep;
    }
  } else {
    // Unaligned-lda path: unconditional scalar loads, still branch-free.
    for (; kt + 32 <= K; kt += 32) {
      float aa[16], cc[16], bb[16];
#pragma unroll
      for (int i = 0; i < 8; ++i) {
        aa[i]     = A0[i];
        aa[i + 8] = A0[16 + i];
        cc[i]     = A1[i];
        cc[i + 8] = A1[16 + i];
        bb[i]     = Wp[(size_t)i * N];
        bb[i + 8] = Wp[(size_t)(i + 16) * N];
      }
      vbf16x16 a = pack16(aa), c = pack16(cc), b = pack16(bb);
      acc0 = __builtin_amdgcn_wmma_f32_16x16x32_bf16(false, a, false, b,
                                                     (short)0, acc0, false, false);
      acc1 = __builtin_amdgcn_wmma_f32_16x16x32_bf16(false, c, false, b,
                                                     (short)0, acc1, false, false);
      A0 += 32; A1 += 32; Wp += wstep;
    }
  }

  if (kt < K) {   // K tail: clamped loads + branchless zero-select on K validity
    float aa[16], cc[16], bb[16];
#pragma unroll
    for (int i = 0; i < 8; ++i) {
      const int k0  = kt + koff + i;          // global k for slot i
      const int k1  = k0 + 16;                // global k for slot i+8
      const int k0c = k0 < K ? k0 : K - 1;
      const int k1c = k1 < K ? k1 : K - 1;
      // signed offsets relative to the advanced bases (== Arow[k0c] etc.)
      const long long d0 = (long long)k0c - kt - koff;
      const long long d1 = (long long)k1c - kt - koff;
      float av0 = A0[d0], av1 = A0[d1];
      float cv0 = A1[d0], cv1 = A1[d1];
      float bv0 = Wp[d0 * N], bv1 = Wp[d1 * N];
      aa[i] = (k0 < K) ? av0 : 0.0f;  aa[i + 8] = (k1 < K) ? av1 : 0.0f;
      cc[i] = (k0 < K) ? cv0 : 0.0f;  cc[i + 8] = (k1 < K) ? cv1 : 0.0f;
      bb[i] = (k0 < K) ? bv0 : 0.0f;  bb[i + 8] = (k1 < K) ? bv1 : 0.0f;
    }
    vbf16x16 a = pack16(aa), c = pack16(cc), b = pack16(bb);
    acc0 = __builtin_amdgcn_wmma_f32_16x16x32_bf16(false, a, false, b,
                                                   (short)0, acc0, false, false);
    acc1 = __builtin_amdgcn_wmma_f32_16x16x32_bf16(false, c, false, b,
                                                   (short)0, acc1, false, false);
  }

  if (colB < N) {
    const float bv = bias ? bias[colB] : 0.0f;
    float* __restrict__ Cp = C + colB;
    const int r0 = tileM + half * 8;        // rows of acc0
    const int r1 = r0 + 16;                 // rows of acc1
    if (tileM + 32 <= M) {                  // interior tile: unguarded stores
#pragma unroll
      for (int i = 0; i < 8; ++i) {
        Cp[(size_t)(r0 + i) * N] = acc0[i] + bv;
        Cp[(size_t)(r1 + i) * N] = acc1[i] + bv;
      }
    } else {
#pragma unroll
      for (int i = 0; i < 8; ++i) {
        if (r0 + i < M) Cp[(size_t)(r0 + i) * N] = acc0[i] + bv;
        if (r1 + i < M) Cp[(size_t)(r1 + i) * N] = acc1[i] + bv;
      }
    }
  }
}

// Per-column batch mean / 1/sqrt(var+eps) (training-mode BN statistics).
__global__ void col_stats(const float* __restrict__ X, int M, int N,
                          float* __restrict__ mean, float* __restrict__ rstd)
{
  __shared__ float sh[512];
  const int col = blockIdx.x;
  if (col >= N) return;
  float s = 0.0f, ss = 0.0f;
  for (int r = threadIdx.x; r < M; r += blockDim.x) {
    float v = X[(size_t)r * N + col];
    s += v; ss += v * v;
  }
  sh[threadIdx.x] = s; sh[256 + threadIdx.x] = ss;
  __syncthreads();
  for (int o = 128; o > 0; o >>= 1) {
    if ((int)threadIdx.x < o) {
      sh[threadIdx.x]       += sh[threadIdx.x + o];
      sh[256 + threadIdx.x] += sh[256 + threadIdx.x + o];
    }
    __syncthreads();
  }
  if (threadIdx.x == 0) {
    float m   = sh[0] / (float)M;
    float var = sh[256] / (float)M - m * m;
    mean[col] = m;
    rstd[col] = rsqrtf(fmaxf(var, 0.0f) + 1e-5f);
  }
}

// In-place: X = act((X-mean)*rstd*g + bt)
__global__ void bn_act(float* __restrict__ X, long long total, int N,
                       const float* __restrict__ mean, const float* __restrict__ rstd,
                       const float* __restrict__ g, const float* __restrict__ bt, int act)
{
  size_t i = (size_t)blockIdx.x * blockDim.x + threadIdx.x;
  if (i >= (size_t)total) return;
  int col = (int)(i % (size_t)N);
  float x = (X[i] - mean[col]) * rstd[col] * g[col] + bt[col];
  X[i] = act_apply(x, act);
}

// hc[e] = concat(h[src[e]], h[dst[e]], hm[e]); hc row stride = ldhc (padded)
__global__ void gather_concat(const float* __restrict__ h, const float* __restrict__ hm,
                              const int* __restrict__ src, const int* __restrict__ dst,
                              float* __restrict__ hc, long long total, int n, int ldhc)
{
  size_t i = (size_t)blockIdx.x * blockDim.x + threadIdx.x;
  if (i >= (size_t)total) return;
  const int n3 = 3 * n;
  size_t e = i / (size_t)n3;
  int j = (int)(i % (size_t)n3);
  float v;
  if (j < n)            v = h[(size_t)src[e] * n + j];
  else if (j < 2 * n)   v = h[(size_t)dst[e] * n + (j - n)];
  else                  v = hm[e * (size_t)n + (j - 2 * n)];
  hc[e * (size_t)ldhc + j] = v;
}

__global__ void mul_inplace(float* __restrict__ a, const float* __restrict__ b,
                            long long total)
{
  size_t i = (size_t)blockIdx.x * blockDim.x + threadIdx.x;
  if (i < (size_t)total) a[i] *= b[i];
}

// agg[dst[e]] += upd[e]   (segment_sum via f32 global atomics)
__global__ void scatter_add(const float* __restrict__ upd, const int* __restrict__ dst,
                            float* __restrict__ agg, long long total, int n)
{
  size_t i = (size_t)blockIdx.x * blockDim.x + threadIdx.x;
  if (i >= (size_t)total) return;
  size_t e = i / (size_t)n;
  int j = (int)(i % (size_t)n);
  atomicAdd(&agg[(size_t)dst[e] * n + j], upd[i]);
}

// h = softplus(BN(agg) + h)
__global__ void bn_add_softplus(const float* __restrict__ agg, float* __restrict__ h,
                                long long total, int N,
                                const float* __restrict__ mean, const float* __restrict__ rstd,
                                const float* __restrict__ g, const float* __restrict__ bt)
{
  size_t i = (size_t)blockIdx.x * blockDim.x + threadIdx.x;
  if (i >= (size_t)total) return;
  int col = (int)(i % (size_t)N);
  float x = (agg[i] - mean[col]) * rstd[col] * g[col] + bt[col] + h[i];
  h[i] = (x > 20.0f) ? x : log1pf(__expf(x));
}

// Graph mean-pool stage 1: atomic sums + counts.
__global__ void pool_accum(const float* __restrict__ h, const int* __restrict__ seg,
                           float* __restrict__ out, float* __restrict__ cnt,
                           long long total, int n, int ostride, int ooff)
{
  size_t i = (size_t)blockIdx.x * blockDim.x + threadIdx.x;
  if (i >= (size_t)total) return;
  size_t r = i / (size_t)n;
  int j = (int)(i % (size_t)n);
  int gidx = seg[r];
  atomicAdd(&out[(size_t)gidx * ostride + ooff + j], h[i]);
  if (j == 0) atomicAdd(&cnt[gidx], 1.0f);
}

__global__ void pool_div(float* __restrict__ out, const float* __restrict__ cnt,
                         int B, int n, int ostride, int ooff)
{
  size_t i = (size_t)blockIdx.x * blockDim.x + threadIdx.x;
  if (i >= (size_t)B * n) return;
  int b = (int)(i / (size_t)n);
  int j = (int)(i % (size_t)n);
  out[(size_t)b * ostride + ooff + j] /= fmaxf(cnt[b], 1.0f);
}

// Row softmax of X[B,n] written into out[:, ooff:ooff+n] (row stride ostride).
__global__ void softmax_out(const float* __restrict__ X, int B, int n,
                            float* __restrict__ out, int ostride, int ooff)
{
  int r = blockIdx.x * blockDim.x + threadIdx.x;
  if (r >= B) return;
  float mx = -3.4e38f;
  for (int j = 0; j < n; ++j) mx = fmaxf(mx, X[(size_t)r * n + j]);
  float s = 0.0f;
  for (int j = 0; j < n; ++j) s += __expf(X[(size_t)r * n + j] - mx);
  float inv = 1.0f / s;
  for (int j = 0; j < n; ++j)
    out[(size_t)r * ostride + ooff + j] = __expf(X[(size_t)r * n + j] - mx) * inv;
}

__global__ void copy_col(const float* __restrict__ x, float* __restrict__ out,
                         int B, int ostride, int ooff)
{
  int r = blockIdx.x * blockDim.x + threadIdx.x;
  if (r < B) out[(size_t)r * ostride + ooff] = x[r];
}

// ---------------------------------------------------------------------------
// Host-side orchestration
// ---------------------------------------------------------------------------

struct MlpP  { const float *W, *b, *bt, *g; };
struct ConvP { const float *bn_b, *bn_g; MlpP edgemlp, mlp, screen; };

static inline unsigned nblk(long long total, int bs) {
  return (unsigned)((total + bs - 1) / bs);
}

// Linear -> BN (batch stats) -> activation.  X has row stride lda (>= Kin).
static void launch_mlp(const float* X, int lda, float* Y, int M, int Kin, int Nout,
                       const MlpP& p, int act, float* mean, float* rstd,
                       hipStream_t s)
{
  dim3 grid((M + 127) / 128, (Nout + 31) / 32);
  gemm_bias_wmma<<<grid, 256, 0, s>>>(X, lda, p.W, p.b, Y, M, Nout, Kin);
  col_stats<<<Nout, 256, 0, s>>>(Y, M, Nout, mean, rstd);
  long long tot = (long long)M * Nout;
  bn_act<<<nblk(tot, 256), 256, 0, s>>>(Y, tot, Nout, mean, rstd, p.g, p.bt, act);
}

static void launch_conv(float* h, const float* ef, const int* src, const int* dst,
                        int Nnode, int Ne, int n, int elen, const ConvP& p,
                        float* hm, float* hc, int ldhc, float* mo, float* so,
                        float* agg, float* mean, float* rstd, hipStream_t s)
{
  // hm = silu(BN(e @ W_edge + b))
  launch_mlp(ef, elen, hm, Ne, elen, n, p.edgemlp, ACT_SILU, mean, rstd, s);
  // hc = [h[src] | h[dst] | hm]  (row stride ldhc, 16B aligned)
  long long tot3 = (long long)Ne * 3 * n;
  gather_concat<<<nblk(tot3, 256), 256, 0, s>>>(h, hm, src, dst, hc, tot3, n, ldhc);
  // upd = silu-MLP(hc) * softplus-MLP(hc)
  launch_mlp(hc, ldhc, mo, Ne, 3 * n, n, p.mlp,    ACT_SILU,     mean, rstd, s);
  launch_mlp(hc, ldhc, so, Ne, 3 * n, n, p.screen, ACT_SOFTPLUS, mean, rstd, s);
  long long tot = (long long)Ne * n;
  mul_inplace<<<nblk(tot, 256), 256, 0, s>>>(mo, so, tot);
  // agg = segment_sum(upd, dst)
  hipMemsetAsync(agg, 0, (size_t)Nnode * n * sizeof(float), s);
  scatter_add<<<nblk(tot, 256), 256, 0, s>>>(mo, dst, agg, tot, n);
  // h = softplus(BN(agg) + h)
  col_stats<<<n, 256, 0, s>>>(agg, Nnode, n, mean, rstd);
  long long totn = (long long)Nnode * n;
  bn_add_softplus<<<nblk(totn, 256), 256, 0, s>>>(agg, h, totn, n, mean, rstd,
                                                  p.bn_g, p.bn_b);
}

extern "C" void kernel_launch(void* const* d_in, const int* in_sizes, int n_in,
                              void* d_out, int out_size, void* d_ws, size_t ws_size,
                              hipStream_t stream)
{
  (void)n_in; (void)out_size; (void)ws_size;

  // ---- unpack params: top-level insertion order, nested pytrees flattened
  //      jax-style (dict keys sorted, 'W' < 'b' < 'bt' < 'g'); 120 leaves. ----
  int c = 0;
  auto nextf = [&]() { return (const float*)d_in[c++]; };
  auto next_mlp = [&]() {
    MlpP m; m.W = nextf(); m.b = nextf(); m.bt = nextf(); m.g = nextf(); return m;
  };

  MlpP atten = next_mlp();                       // 'atten'
  const float* atten_bn_b = nextf();             // 'atten_bn_b'
  const float* atten_bn_g = nextf();             // 'atten_bn_g'
  ConvP conv_a[3], conv_s[3];
  for (int i = 0; i < 3; ++i) {                  // 'conv_a' (keys: bn_b,bn_g,edgemlp,mlp,screen)
    conv_a[i].bn_b = nextf(); conv_a[i].bn_g = nextf();
    conv_a[i].edgemlp = next_mlp(); conv_a[i].mlp = next_mlp(); conv_a[i].screen = next_mlp();
  }
  for (int i = 0; i < 3; ++i) {                  // 'conv_s'
    conv_s[i].bn_b = nextf(); conv_s[i].bn_g = nextf();
    conv_s[i].edgemlp = next_mlp(); conv_s[i].mlp = next_mlp(); conv_s[i].screen = next_mlp();
  }
  MlpP emb_a = next_mlp();                       // 'emb_a'
  MlpP emb_s = next_mlp();                       // 'emb_s'
  MlpP fc[3]; for (int i = 0; i < 3; ++i) fc[i] = next_mlp();   // 'fc'
  MlpP pred_adsb = next_mlp();                   // 'pred_adsb'
  MlpP pred_site = next_mlp();                   // 'pred_site'
  const float* predt_W = nextf();                // 'pred_t' {W,b}
  const float* predt_b = nextf();

  // ---- data arrays ----
  const int va_i = c;
  const float* va = nextf();
  const float* ea = nextf();
  const float* vs = nextf();
  const float* es = nextf();
  const int* src_a = (const int*)d_in[c++];
  const int* dst_a = (const int*)d_in[c++];
  const int* n2g_a = (const int*)d_in[c++];
  const int* src_s = (const int*)d_in[c++];
  const int* dst_s = (const int*)d_in[c++];
  const int* n2g_s = (const int*)d_in[c++];

  const int Na = in_sizes[va_i + 0] / 101;
  const int Ea = in_sizes[va_i + 1] / 6;
  const int Ns = in_sizes[va_i + 2] / 101;
  const int Es = in_sizes[va_i + 3] / 8;
  const int B  = 512;
  const int NA = 110, NS = 150;              // branch widths
  const int LDA3 = (3 * NA + 3) & ~3;        // 330 -> 332 (16B-aligned rows)
  const int LDS3 = (3 * NS + 3) & ~3;        // 450 -> 452

  // ---- workspace bump allocator (256B-aligned chunks) ----
  float* wsf = (float*)d_ws;
  size_t off = 0;
  auto alloc = [&](size_t nelem) {
    float* p = wsf + off;
    off += (nelem + 63) & ~(size_t)63;
    return p;
  };
  auto maxz = [](size_t a, size_t b) { return a > b ? a : b; };

  float* hbuf = alloc(maxz((size_t)Na * NA, (size_t)Ns * NS));
  float* hm   = alloc(maxz((size_t)Ea * NA, (size_t)Es * NS));
  float* hc   = alloc(maxz((size_t)Ea * LDA3, (size_t)Es * LDS3));
  float* mo   = alloc(maxz((size_t)Ea * NA, (size_t)Es * NS));
  float* so   = alloc(maxz((size_t)Ea * NA, (size_t)Es * NS));
  float* agg  = alloc(maxz((size_t)Na * NA, (size_t)Ns * NS));
  float* vt   = alloc((size_t)B * 260);
  float* t1   = alloc((size_t)B * 260);
  float* fb0  = alloc((size_t)B * 80);
  float* fb1  = alloc((size_t)B * 120);
  float* fb2  = alloc((size_t)B * 10);
  float* ca   = alloc((size_t)B * 5);
  float* cs   = alloc((size_t)B * 4);
  float* tg   = alloc((size_t)B);
  float* cnt  = alloc(512);
  float* mean = alloc(512);
  float* rstd = alloc(512);

  float* out = (float*)d_out;

  // ================= adsorbate branch =================
  launch_mlp(va, 101, hbuf, Na, 101, NA, emb_a, ACT_SILU, mean, rstd, stream);
  for (int i = 0; i < 3; ++i)
    launch_conv(hbuf, ea, src_a, dst_a, Na, Ea, NA, 6, conv_a[i],
                hm, hc, LDA3, mo, so, agg, mean, rstd, stream);
  // mean-pool into vt[:, 0:110]
  hipMemsetAsync(vt, 0, (size_t)B * 260 * sizeof(float), stream);
  hipMemsetAsync(cnt, 0, (size_t)B * sizeof(float), stream);
  {
    long long tot = (long long)Na * NA;
    pool_accum<<<nblk(tot, 256), 256, 0, stream>>>(hbuf, n2g_a, vt, cnt, tot, NA, 260, 0);
    pool_div<<<nblk((long long)B * NA, 256), 256, 0, stream>>>(vt, cnt, B, NA, 260, 0);
  }

  // ================= slab branch =================
  launch_mlp(vs, 101, hbuf, Ns, 101, NS, emb_s, ACT_SILU, mean, rstd, stream);
  for (int i = 0; i < 3; ++i)
    launch_conv(hbuf, es, src_s, dst_s, Ns, Es, NS, 8, conv_s[i],
                hm, hc, LDS3, mo, so, agg, mean, rstd, stream);
  // mean-pool into vt[:, 110:260]
  hipMemsetAsync(cnt, 0, (size_t)B * sizeof(float), stream);
  {
    long long tot = (long long)Ns * NS;
    pool_accum<<<nblk(tot, 256), 256, 0, stream>>>(hbuf, n2g_s, vt, cnt, tot, NS, 260, NA);
    pool_div<<<nblk((long long)B * NS, 256), 256, 0, stream>>>(vt, cnt, B, NS, 260, NA);
  }

  // ================= attention gate =================
  // vt = BN( elu-MLP(vt) * vt , atten_bn_g, atten_bn_b )
  launch_mlp(vt, 260, t1, B, 260, 260, atten, ACT_ELU, mean, rstd, stream);
  mul_inplace<<<nblk((long long)B * 260, 256), 256, 0, stream>>>(vt, t1, (long long)B * 260);
  col_stats<<<260, 256, 0, stream>>>(vt, B, 260, mean, rstd);
  bn_act<<<nblk((long long)B * 260, 256), 256, 0, stream>>>(
      vt, (long long)B * 260, 260, mean, rstd, atten_bn_g, atten_bn_b, ACT_NONE);

  // ================= FC stack =================
  launch_mlp(vt,  260, fb0, B, 260, 80,  fc[0], ACT_SILU, mean, rstd, stream);
  launch_mlp(fb0, 80,  fb1, B, 80,  120, fc[1], ACT_SILU, mean, rstd, stream);
  launch_mlp(fb1, 120, fb2, B, 120, 10,  fc[2], ACT_SILU, mean, rstd, stream);

  // ================= heads =================
  launch_mlp(fb2, 10, ca, B, 10, 5, pred_adsb, ACT_NONE, mean, rstd, stream);
  softmax_out<<<nblk(B, 256), 256, 0, stream>>>(ca, B, 5, out, 10, 0);
  launch_mlp(fb2, 10, cs, B, 10, 4, pred_site, ACT_NONE, mean, rstd, stream);
  softmax_out<<<nblk(B, 256), 256, 0, stream>>>(cs, B, 4, out, 10, 5);
  {
    dim3 grid((B + 127) / 128, 1);
    gemm_bias_wmma<<<grid, 256, 0, stream>>>(fb2, 10, predt_W, predt_b, tg, B, 1, 10);
    copy_col<<<nblk(B, 256), 256, 0, stream>>>(tg, out, B, 10, 9);
  }
}
